// Transfusion_47648367182407
// MI455X (gfx1250) — matrix operand order repaired and checked by reading, not compiled, not run on hardware
//
#include <hip/hip_runtime.h>
#include <hip/hip_bf16.h>
#include <math.h>

typedef __attribute__((ext_vector_type(16))) _Float16 v16h;
typedef __attribute__((ext_vector_type(8)))  float    v8f;

#define B_   2
#define N_   2048
#define D_   1024
#define H_   16
#define DH_  64
#define SOFTCAP_ 50.0f

// ---------------------------------------------------------------------------
// CDNA5 async global->LDS copy (CDNA5 ISA):
// per-lane: copies 16B from global memory into LDS, counted by ASYNCcnt.
// INST_OFFSET is added to BOTH sides, so consecutive copies of a row share one
// base register pair and differ only in the immediate. LDS byte address = low
// 32 bits of the generic __shared__ pointer.
// ---------------------------------------------------------------------------
template<int OFF>
__device__ inline void async_b128_o(uint32_t lds_off, const void* g) {
  asm volatile("global_load_async_to_lds_b128 %0, %1, off offset:%2"
               :: "v"(lds_off), "v"(g), "n"(OFF)
               : "memory");
}
// COUNT consecutive 16-byte copies (same row on both sides).
template<int COUNT>
__device__ inline void async_copy_row(uint32_t lds_off, const void* g) {
  if constexpr (COUNT > 0) {
    async_copy_row<COUNT - 1>(lds_off, g);
    async_b128_o<(COUNT - 1) * 16>(lds_off, g);
  }
}
// Wait until at most N async ops remain outstanding (in-order completion).
template<int N>
__device__ inline void wait_async() {
  asm volatile("s_wait_asynccnt %0" :: "n"(N) : "memory");
}
__device__ inline uint32_t lds_off_of(const void* p) {
  return (uint32_t)(uintptr_t)p;
}

// ---------------------------------------------------------------------------
// WMMA fragment helpers (CDNA5 wave32 layouts, CDNA5 ISA §7.12.2)
// A 16x32 f16: lane L: M=L&15, lanes0-15 K=0..15, lanes16-31 K=16..31 (contiguous, as B)
// B 32x16 f16 from K-major tile bt[n][k]: lane L: N=L&15, K half by lane>=16
// B 32x16 f16 from row-major tile b[k][n]: element e -> K = kb + e
// C/D 16x16 f32: lane L: N=L&15, VGPR r -> M = r + (L>=16 ? 8 : 0)
// ---------------------------------------------------------------------------
__device__ inline v16h lds_load_a_frag(const _Float16* tile, int ldk) {
  int lane = threadIdx.x & 31;
  int m    = lane & 15;
  int kb2  = (lane & 16) ? 8 : 0;           // kb/2 in dwords
  const uint32_t* r32 = (const uint32_t*)(tile + m * ldk);
  v16h a;
  uint32_t* a32 = (uint32_t*)&a;
#pragma unroll
  for (int i = 0; i < 4; ++i) a32[i]     = r32[kb2 + i];        // K 0..7 / 16..23
#pragma unroll
  for (int i = 0; i < 4; ++i) a32[4 + i] = r32[4 + kb2 + i];    // K 8..15 / 24..31
  return a;
}

__device__ inline v16h lds_load_bt_frag(const _Float16* bt, int ldk) {
  int lane = threadIdx.x & 31;
  int n    = lane & 15;
  int kb2  = (lane & 16) ? 8 : 0;           // K 0..15 or 16..31 (dwords)
  const uint32_t* r32 = (const uint32_t*)(bt + n * ldk);
  v16h b;
  uint32_t* b32 = (uint32_t*)&b;
#pragma unroll
  for (int i = 0; i < 8; ++i) b32[i] = r32[kb2 + i];
  return b;
}

__device__ inline v16h lds_load_b_gather_frag(const _Float16* brow, int ldn) {
  int lane = threadIdx.x & 31;
  int n    = lane & 15;
  int kb   = (lane & 16) ? 16 : 0;
  v16h b;
  _Float16* bh = (_Float16*)&b;
#pragma unroll
  for (int e = 0; e < 16; ++e) bh[e] = brow[(kb + e) * ldn + n];
  return b;
}

__device__ inline v8f wmma_f16(v16h a, v16h b, v8f c) {
  return __builtin_amdgcn_wmma_f32_16x16x32_f16(false, a, false, b, (short)0, c, false, false);
}

// ---------------------------------------------------------------------------
// f32 -> f16 weight conversion
// ---------------------------------------------------------------------------
__global__ __launch_bounds__(256)
void f32_to_f16_kernel(const float* __restrict__ in, _Float16* __restrict__ out, int n) {
  int i = blockIdx.x * blockDim.x + threadIdx.x;
  if (i < n) out[i] = (_Float16)in[i];
}

// ---------------------------------------------------------------------------
// RoPE positions: pos[b][n] = n - inclusive_cumsum(is_any_modality)
// ---------------------------------------------------------------------------
__global__ __launch_bounds__(128)
void pos_kernel(const long long* __restrict__ mods, float* __restrict__ pos) {
  __shared__ int partial[128];
  int b = blockIdx.x;
  int off[4], len[4];
#pragma unroll
  for (int m = 0; m < 4; ++m) {
    off[m] = (int)mods[b * 12 + m * 3 + 1];
    len[m] = (int)mods[b * 12 + m * 3 + 2];
  }
  int t = threadIdx.x;
  int base = t * 16;
  unsigned flags = 0;
  int cnt = 0;
#pragma unroll
  for (int e = 0; e < 16; ++e) {
    int n = base + e;
    int f = 0;
#pragma unroll
    for (int m = 0; m < 4; ++m)
      f |= (n >= off[m] + 1 && n < off[m] + len[m]) ? 1 : 0;
    flags |= (unsigned)f << e;
    cnt += f;
  }
  partial[t] = cnt;
  __syncthreads();
  if (t == 0) {
    int run = 0;
    for (int i = 0; i < 128; ++i) { int c = partial[i]; partial[i] = run; run += c; }
  }
  __syncthreads();
  int run = partial[t];
#pragma unroll
  for (int e = 0; e < 16; ++e) {
    int n = base + e;
    run += (flags >> e) & 1;
    pos[(size_t)b * N_ + n] = (float)(n - run);
  }
}

// ---------------------------------------------------------------------------
// L2-norm scale: xn = x / max(||x||,1e-12) * sqrt(D) * (gamma + 1)  -> f16
// ---------------------------------------------------------------------------
__global__ __launch_bounds__(256)
void rmsnorm_kernel(const float* __restrict__ x, const float* __restrict__ gamma,
                    _Float16* __restrict__ xn) {
  __shared__ float red[256];
  int row = blockIdx.x;                       // b*N + n
  const float* xr = x + (size_t)row * D_;
  float xs[4];
  float s = 0.f;
#pragma unroll
  for (int t = 0; t < 4; ++t) {
    xs[t] = xr[threadIdx.x + t * 256];
    s += xs[t] * xs[t];
  }
  red[threadIdx.x] = s;
  __syncthreads();
  for (int d = 128; d > 0; d >>= 1) {
    if (threadIdx.x < d) red[threadIdx.x] += red[threadIdx.x + d];
    __syncthreads();
  }
  float scale = 32.0f / fmaxf(sqrtf(red[0]), 1e-12f);   // sqrt(1024)=32
#pragma unroll
  for (int t = 0; t < 4; ++t) {
    int c = threadIdx.x + t * 256;
    xn[(size_t)row * D_ + c] = (_Float16)(xs[t] * scale * (gamma[c] + 1.0f));
  }
}

// ---------------------------------------------------------------------------
// WMMA GEMM: C[M,N] = A[M,K] * B[K,N], f16 in, f32 acc.
// Block tile 128x64, 4 waves: wave w owns rows [w*32, w*32+32) x all 64 cols.
// Per k-step (BK=32): 2 A-frags x 4 B-frags = 8 WMMA per wave.
// Async double-buffered staging: issue tile kt+32 into buf 1-p, then
// s_wait_asynccnt 6 (per-wave ops of the next tile) => current tile resident.
// ---------------------------------------------------------------------------
template<int HALF_OUT>
__global__ __launch_bounds__(128)
void gemm_wmma_kernel(const _Float16* __restrict__ A, int lda,
                      const _Float16* __restrict__ Bm, int ldb,
                      void* __restrict__ Cv, int ldc, int K) {
  __shared__ _Float16 sA[2][128 * 32];    // [m][k] row-major
  __shared__ _Float16 sB[2][32 * 64];     // [k][n] row-major
  const int tid  = threadIdx.x;
  const int wave = tid >> 5;
  const int wm   = wave * 32;
  const int m0   = blockIdx.y * 128;
  const int n0   = blockIdx.x * 64;
  uint32_t sA_off[2] = { lds_off_of(sA[0]), lds_off_of(sA[1]) };
  uint32_t sB_off[2] = { lds_off_of(sB[0]), lds_off_of(sB[1]) };
  v8f acc[2][4] = {};

  // thread t stages: A row t (4 consecutive b128), B row t>>2, cols (t&3)*16..+15
  const uint32_t aL = (uint32_t)tid * 64;                      // bytes into sA
  const uint32_t bL = (uint32_t)((tid >> 2) * 128 + (tid & 3) * 32);
  const _Float16* aG0 = A  + (size_t)(m0 + tid) * lda;
  const _Float16* bG0 = Bm + (size_t)(tid >> 2) * ldb + n0 + (tid & 3) * 16;

  auto stage = [&](int kt, int pb) {
    async_copy_row<4>(sA_off[pb] + aL, aG0 + kt);
    async_copy_row<2>(sB_off[pb] + bL, bG0 + (size_t)kt * ldb);
  };

  stage(0, 0);
  int p = 0;
  for (int kt = 0; kt < K; kt += 32) {
    if (kt + 32 < K) { stage(kt + 32, p ^ 1); wait_async<6>(); }
    else             { wait_async<0>(); }
    __syncthreads();

    v16h aF[2], bF[4];
#pragma unroll
    for (int i = 0; i < 2; ++i) aF[i] = lds_load_a_frag(sA[p] + (wm + i * 16) * 32, 32);
#pragma unroll
    for (int j = 0; j < 4; ++j) bF[j] = lds_load_b_gather_frag(sB[p] + j * 16, 64);
#pragma unroll
    for (int i = 0; i < 2; ++i)
#pragma unroll
      for (int j = 0; j < 4; ++j)
        acc[i][j] = wmma_f16(aF[i], bF[j], acc[i][j]);
    __syncthreads();
    p ^= 1;
  }

  const int lane = tid & 31;
  const int nl   = lane & 15;
  const int radd = (lane & 16) ? 8 : 0;
#pragma unroll
  for (int i = 0; i < 2; ++i)
#pragma unroll
    for (int j = 0; j < 4; ++j)
#pragma unroll
      for (int r = 0; r < 8; ++r) {
        size_t gm = (size_t)(m0 + wm + i * 16 + r + radd);
        size_t gn = (size_t)(n0 + j * 16 + nl);
        float v = acc[i][j][r];
        if (HALF_OUT) ((_Float16*)Cv)[gm * ldc + gn] = (_Float16)v;
        else          ((float*)Cv)[gm * ldc + gn]    = v;
      }
}

// ---------------------------------------------------------------------------
// RoPE + QKV split: qkv[bn][s*1024 + h*64 + d] -> q/k/v[b][h][n][d] (f16)
// q is pre-scaled by dh^-0.5 = 0.125; softcap /50 applied in attention.
// ---------------------------------------------------------------------------
__global__ __launch_bounds__(256)
void rope_split_kernel(const _Float16* __restrict__ qkv, const float* __restrict__ pos,
                       _Float16* __restrict__ qo, _Float16* __restrict__ ko,
                       _Float16* __restrict__ vo) {
  int bn = blockIdx.x;
  int b = bn / N_, n = bn % N_;
  float p = pos[bn];
  const _Float16* row = qkv + (size_t)bn * 3072;
  for (int pp = threadIdx.x; pp < 1024; pp += 256) {
    int s   = pp >> 9;          // 0 = q, 1 = k
    int rem = pp & 511;
    int h = rem >> 5, i = rem & 31;
    float freq = __powf(10000.0f, -(float)(2 * i) * (1.0f / 64.0f));
    float ang = p * freq;
    float cs = __cosf(ang), sn = __sinf(ang);
    float x0 = (float)row[s * 1024 + h * 64 + 2 * i];
    float x1 = (float)row[s * 1024 + h * 64 + 2 * i + 1];
    float scl = (s == 0) ? 0.125f : 1.0f;
    float y0 = (x0 * cs - x1 * sn) * scl;
    float y1 = (x1 * cs + x0 * sn) * scl;
    _Float16* dst = (s == 0 ? qo : ko) + ((size_t)(b * H_ + h) * N_ + n) * DH_ + 2 * i;
    dst[0] = (_Float16)y0;
    dst[1] = (_Float16)y1;
  }
  for (int c = threadIdx.x; c < 1024; c += 256) {
    int h = c >> 6, d = c & 63;
    vo[((size_t)(b * H_ + h) * N_ + n) * DH_ + d] = row[2048 + c];
  }
}

// ---------------------------------------------------------------------------
// Flash attention, 1 wave / 16 query rows, 32-key chunks, WMMA QK^T and PV.
// logits = tanh(s * 0.02) * 50, masked by causal | modality windows.
// K/V double-buffered via async global->LDS with partial asynccnt waits.
// ---------------------------------------------------------------------------
__global__ __launch_bounds__(32)
void attn_kernel(const _Float16* __restrict__ qg, const _Float16* __restrict__ kg,
                 const _Float16* __restrict__ vg, const long long* __restrict__ mods,
                 _Float16* __restrict__ outg) {
  __shared__ _Float16 qt[16 * 64];       // Q tile  [m][d]
  __shared__ _Float16 kt[2][32 * 64];    // K tile  [j][d]  (B^T layout for QK^T)
  __shared__ _Float16 vt[2][32 * 64];    // V tile  [j][d]  (PV uses gather frag)
  __shared__ _Float16 pt[16 * 32];       // P tile  [m][j]  (A layout source for PV)

  const int lane = threadIdx.x;
  const int m0 = blockIdx.x * 16;
  const int h  = blockIdx.y;
  const int b  = blockIdx.z;
  const uint32_t qt_off = lds_off_of(qt);
  uint32_t kt_off[2] = { lds_off_of(kt[0]), lds_off_of(kt[1]) };
  uint32_t vt_off[2] = { lds_off_of(vt[0]), lds_off_of(vt[1]) };

  int offm[4], endm[4];
#pragma unroll
  for (int m = 0; m < 4; ++m) {
    long long o = mods[b * 12 + m * 3 + 1];
    long long L = mods[b * 12 + m * 3 + 2];
    offm[m] = (int)o;
    endm[m] = (int)(o + L);
  }
  int jlimit = m0 + 16;
#pragma unroll
  for (int m = 0; m < 4; ++m)
    if (m0 + 15 >= offm[m] && endm[m] > jlimit) jlimit = endm[m];
  int jmax = (jlimit + 31) & ~31;
  if (jmax > N_) jmax = N_;

  const size_t headbase = (size_t)(b * H_ + h) * N_;
  // stage Q: lane copies 64 consecutive bytes (4 b128)
  async_copy_row<4>(qt_off + lane * 64, (const char*)(qg + (headbase + m0) * DH_) + lane * 64);

  // stage K/V chunk: one 128-byte row per lane, 8 b128 each
  auto stage_kv = [&](int j0, int pb) {
    const _Float16* kr = kg + (headbase + j0 + lane) * DH_;
    const _Float16* vr = vg + (headbase + j0 + lane) * DH_;
    async_copy_row<8>(kt_off[pb] + lane * 128, kr);
    async_copy_row<8>(vt_off[pb] + lane * 128, vr);
  };

  stage_kv(0, 0);
  wait_async<16>();                 // Q resident (kv chunk 0 may still be in flight)
  __syncthreads();
  v16h aQ0 = lds_load_a_frag(qt, 64);        // K dims 0..31
  v16h aQ1 = lds_load_a_frag(qt + 32, 64);   // K dims 32..63

  v8f o[4] = {};
  float mstate[8], lstate[8];
#pragma unroll
  for (int r = 0; r < 8; ++r) { mstate[r] = -1e30f; lstate[r] = 0.f; }

  const int nl   = lane & 15;
  const int radd = (lane & 16) ? 8 : 0;

  int p = 0;
  for (int j0 = 0; j0 < jmax; j0 += 32) {
    if (j0 + 32 < jmax) { stage_kv(j0 + 32, p ^ 1); wait_async<16>(); }
    else                { wait_async<0>(); }
    __syncthreads();

    // S tiles: 16 rows x 32 keys, contraction over dh=64 (2 WMMA per tile)
    v8f s0 = {}, s1 = {};
    s0 = wmma_f16(aQ0, lds_load_bt_frag(kt[p], 64), s0);
    s0 = wmma_f16(aQ1, lds_load_bt_frag(kt[p] + 32, 64), s0);
    s1 = wmma_f16(aQ0, lds_load_bt_frag(kt[p] + 16 * 64, 64), s1);
    s1 = wmma_f16(aQ1, lds_load_bt_frag(kt[p] + 16 * 64 + 32, 64), s1);

    float alpha_r[8];
#pragma unroll
    for (int r = 0; r < 8; ++r) {
      int i  = m0 + r + radd;
      int ja = j0 + nl;
      int jb = j0 + 16 + nl;
      float v0 = tanhf(s0[r] * 0.02f) * SOFTCAP_;   // scale/50 folded: q*0.125, *0.02
      float v1 = tanhf(s1[r] * 0.02f) * SOFTCAP_;
      bool a0 = (i >= ja), a1 = (i >= jb);
#pragma unroll
      for (int m = 0; m < 4; ++m) {
        bool rowin = (i >= offm[m]);
        a0 = a0 || (rowin && (ja < endm[m]));
        a1 = a1 || (rowin && (jb < endm[m]));
      }
      v0 = a0 ? v0 : -1e30f;
      v1 = a1 ? v1 : -1e30f;
      float mx = fmaxf(v0, v1);
#pragma unroll
      for (int d = 1; d < 16; d <<= 1) mx = fmaxf(mx, __shfl_xor(mx, d, 32));
      float mnew = fmaxf(mstate[r], mx);
      float al = __expf(mstate[r] - mnew);
      float p0 = __expf(v0 - mnew);
      float p1 = __expf(v1 - mnew);
      pt[(r + radd) * 32 + nl]      = (_Float16)p0;
      pt[(r + radd) * 32 + 16 + nl] = (_Float16)p1;
      float ps = p0 + p1;
#pragma unroll
      for (int d = 1; d < 16; d <<= 1) ps += __shfl_xor(ps, d, 32);
      lstate[r] = lstate[r] * al + ps;
      mstate[r] = mnew;
      alpha_r[r] = al;
    }
#pragma unroll
    for (int c = 0; c < 4; ++c)
#pragma unroll
      for (int r = 0; r < 8; ++r) o[c][r] *= alpha_r[r];
    __syncthreads();

    // O += P(16x32) @ V(32x64), V consumed row-major via K-gather fragments
    v16h aP = lds_load_a_frag(pt, 32);
#pragma unroll
    for (int c = 0; c < 4; ++c)
      o[c] = wmma_f16(aP, lds_load_b_gather_frag(vt[p] + c * 16, 64), o[c]);
    __syncthreads();
    p ^= 1;
  }

  // normalize + store to [b][n][h*64 + d] (f16) for the output GEMM
#pragma unroll
  for (int c = 0; c < 4; ++c)
#pragma unroll
    for (int r = 0; r < 8; ++r) {
      int i   = m0 + r + radd;
      int col = h * DH_ + c * 16 + nl;
      float v = o[c][r] / lstate[r];
      outg[(size_t)(b * N_ + i) * D_ + col] = (_Float16)v;
    }
}

// ---------------------------------------------------------------------------
// Launch
// ---------------------------------------------------------------------------
extern "C" void kernel_launch(void* const* d_in, const int* in_sizes, int n_in,
                              void* d_out, int out_size, void* d_ws, size_t ws_size,
                              hipStream_t stream) {
  (void)in_sizes; (void)n_in; (void)out_size; (void)ws_size;
  const float*     x     = (const float*)d_in[0];
  const long long* mods  = (const long long*)d_in[1];
  const float*     gamma = (const float*)d_in[2];
  const float*     w_qkv = (const float*)d_in[3];
  const float*     w_out = (const float*)d_in[4];
  float*           out   = (float*)d_out;

  char* ws = (char*)d_ws;
  size_t off = 0;
  auto take = [&](size_t bytes) -> char* {
    off = (off + 255) & ~(size_t)255;
    char* p = ws + off;
    off += bytes;
    return p;
  };
  float*    pos    = (float*)   take((size_t)B_ * N_ * 4);
  _Float16* xn     = (_Float16*)take((size_t)B_ * N_ * D_ * 2);
  _Float16* wqkv_h = (_Float16*)take((size_t)D_ * 3072 * 2);
  _Float16* wout_h = (_Float16*)take((size_t)D_ * D_ * 2);
  _Float16* qkv    = (_Float16*)take((size_t)B_ * N_ * 3072 * 2);
  _Float16* qb     = (_Float16*)take((size_t)B_ * H_ * N_ * DH_ * 2);
  _Float16* kb     = (_Float16*)take((size_t)B_ * H_ * N_ * DH_ * 2);
  _Float16* vb     = (_Float16*)take((size_t)B_ * H_ * N_ * DH_ * 2);
  _Float16* attn   = (_Float16*)take((size_t)B_ * N_ * D_ * 2);

  f32_to_f16_kernel<<<(D_ * 3072 + 255) / 256, 256, 0, stream>>>(w_qkv, wqkv_h, D_ * 3072);
  f32_to_f16_kernel<<<(D_ * D_ + 255) / 256, 256, 0, stream>>>(w_out, wout_h, D_ * D_);
  pos_kernel<<<B_, 128, 0, stream>>>(mods, pos);
  rmsnorm_kernel<<<B_ * N_, 256, 0, stream>>>(x, gamma, xn);
  gemm_wmma_kernel<1><<<dim3(3072 / 64, (B_ * N_) / 128), 128, 0, stream>>>(
      xn, D_, wqkv_h, 3072, (void*)qkv, 3072, D_);
  rope_split_kernel<<<B_ * N_, 256, 0, stream>>>(qkv, pos, qb, kb, vb);
  attn_kernel<<<dim3(N_ / 16, H_, B_), 32, 0, stream>>>(qb, kb, vb, mods, attn);
  gemm_wmma_kernel<0><<<dim3(D_ / 64, (B_ * N_) / 128), 128, 0, stream>>>(
      attn, D_, wout_h, D_, (void*)out, D_, D_);
}
